// MotionModelCont_86354612453596
// MI455X (gfx1250) — matrix operand order repaired and checked by reading, not compiled, not run on hardware
//
#include <hip/hip_runtime.h>
#include <math.h>

// ---------------------------------------------------------------------------
// Model dims
// ---------------------------------------------------------------------------
#define V_  32000
#define H_  1024
#define NH_ 16
#define HD_ 64
#define F_  4096
#define NL_ 4
#define R_  32
#define D_  263
#define B_  4
#define LT_ 64
#define T_  196
#define S_  260          // LT_ + T_
#define M_  (B_*S_)      // 1040 token rows
#define BT_ (B_*T_)      // 784 motion rows
#define TLROWS_ (B_*(LT_-1))  // 252 language-loss rows
#define LORA_SCALE_ 2.0f

// ---------------------------------------------------------------------------
// WMMA types (gfx1250, wave32)
// ---------------------------------------------------------------------------
typedef __attribute__((ext_vector_type(16))) __bf16       v16bf;
typedef __attribute__((ext_vector_type(8)))  float        v8f;
typedef __attribute__((ext_vector_type(8)))  unsigned int v8u;
typedef __attribute__((ext_vector_type(4)))  float        v4f;

__device__ __forceinline__ unsigned int f2bf(float f) {
    unsigned int u = __builtin_bit_cast(unsigned int, f);
    unsigned int r = u + 0x7FFFu + ((u >> 16) & 1u);   // round-to-nearest-even
    return r >> 16;
}
__device__ __forceinline__ unsigned int pack_bf2(float a, float b) {
    return f2bf(a) | (f2bf(b) << 16);
}
__device__ __forceinline__ float gelu_exact(float x) {
    return 0.5f * x * (1.0f + erff(x * 0.70710678118654752f));
}
__device__ __forceinline__ float silu_f(float x) { return x / (1.0f + expf(-x)); }
__device__ __forceinline__ float sigmoid_f(float x) { return 1.0f / (1.0f + expf(-x)); }
__device__ __forceinline__ float softplus_f(float x) {
    return fmaxf(x, 0.0f) + log1pf(expf(-fabsf(x)));
}

// ---------------------------------------------------------------------------
// Generic bf16-WMMA GEMM:  C[M,N] (op)= act( (A[M,K] @ W[N,K]^T) * scale + bias )
//   accumulate==0 : C = act(acc*scale + bias)
//   accumulate==1 : C += acc*scale            (bias/act ignored)
// Block: 256 threads (8 waves). Tile 128(M) x 64(N) x 64(K).
// Each wave owns a 32x32 patch -> 8 x v_wmma_f32_16x16x32_bf16 per K tile.
// Interior tiles stage with unguarded global_load_b128; edges use the guarded
// scalar path. Next K tile is prefetched (global_prefetch_b8).
// ---------------------------------------------------------------------------
#define GTM 128
#define GTN 64
#define GTK 64
#define LDSPITCH (GTK/2 + 1)   // 33 dwords per row

__global__ __launch_bounds__(256) void gemm_bf16_kernel(
    const float* __restrict__ A, const float* __restrict__ W, float* __restrict__ C,
    int M, int N, int K, int lda, int ldw, int ldc,
    const float* __restrict__ bias, float scale, int accumulate, int act)
{
    __shared__ unsigned int uA[GTM][LDSPITCH];   // K packed as bf16 pairs
    __shared__ unsigned int uB[GTN][LDSPITCH];

    const int tid  = threadIdx.x;
    const int lane = tid & 31;
    const int wave = tid >> 5;       // 0..7
    const int wm   = wave & 3;       // 32-row group within 128
    const int wn   = wave >> 2;      // 32-col group within 64
    const int m0   = blockIdx.y * GTM;
    const int n0   = blockIdx.x * GTN;

    const int lmA = lane & 15;       // row/col within 16-tile
    const int hiA = lane >> 4;       // lane half

    const bool alignedA = ((lda & 3) == 0);
    const bool alignedB = ((ldw & 3) == 0);
    const bool fullM = (m0 + GTM <= M);
    const bool fullN = (n0 + GTN <= N);

    v8f c00 = {}; v8f c01 = {}; v8f c10 = {}; v8f c11 = {};

    for (int k0 = 0; k0 < K; k0 += GTK) {
        const bool fullK = (k0 + GTK <= K);

        // ---- stage A tile (128 x 64 f32 -> bf16 pairs) ----
        if (fullM && fullK && alignedA) {
            // 128 rows x 16 float4 per row = 2048 vector loads / 256 threads
            #pragma unroll
            for (int it = 0; it < 8; ++it) {
                int t = tid + it * 256;
                int m = t >> 4, q4 = t & 15;
                const v4f v = *(const v4f*)(A + (size_t)(m0 + m)*lda + k0 + q4*4);
                uA[m][q4*2 + 0] = pack_bf2(v.x, v.y);
                uA[m][q4*2 + 1] = pack_bf2(v.z, v.w);
            }
        } else {
            for (int t = tid; t < GTM * (GTK/2); t += 256) {
                int m = t >> 5, p = t & 31;
                int gm = m0 + m, gk = k0 + 2*p;
                float a0 = (gm < M && gk     < K) ? A[(size_t)gm*lda + gk]     : 0.0f;
                float a1 = (gm < M && gk + 1 < K) ? A[(size_t)gm*lda + gk + 1] : 0.0f;
                uA[m][p] = pack_bf2(a0, a1);
            }
        }
        // ---- stage W tile (64 x 64) ----
        if (fullN && fullK && alignedB) {
            #pragma unroll
            for (int it = 0; it < 4; ++it) {
                int t = tid + it * 256;
                int n = t >> 4, q4 = t & 15;
                const v4f v = *(const v4f*)(W + (size_t)(n0 + n)*ldw + k0 + q4*4);
                uB[n][q4*2 + 0] = pack_bf2(v.x, v.y);
                uB[n][q4*2 + 1] = pack_bf2(v.z, v.w);
            }
        } else {
            for (int t = tid; t < GTN * (GTK/2); t += 256) {
                int n = t >> 5, p = t & 31;
                int gn = n0 + n, gk = k0 + 2*p;
                float b0 = (gn < N && gk     < K) ? W[(size_t)gn*ldw + gk]     : 0.0f;
                float b1 = (gn < N && gk + 1 < K) ? W[(size_t)gn*ldw + gk + 1] : 0.0f;
                uB[n][p] = pack_bf2(b0, b1);
            }
        }

        // ---- prefetch next K tile while this one is consumed ----
        if (k0 + GTK < K) {
            int pm = m0 + (tid & 127);
            if (pm < M) __builtin_prefetch(A + (size_t)pm*lda + k0 + GTK, 0, 3);
            if (tid < GTN && n0 + tid < N)
                __builtin_prefetch(W + (size_t)(n0 + tid)*ldw + k0 + GTK, 0, 3);
        }
        __syncthreads();

        // ---- 2 WMMA sub-steps of K=32 over the staged K=64 tile ----
        #pragma unroll
        for (int ks = 0; ks < 2; ++ks) {
            v8u au0, au1, bu0, bu1;
            #pragma unroll
            for (int r = 0; r < 8; ++r) {
                int pa = ks*16 + ((r < 4) ? 0 : 8) + hiA*4 + (r & 3);  // ISA A layout
                au0[r] = uA[wm*32 +      lmA][pa];
                au1[r] = uA[wm*32 + 16 + lmA][pa];
                int pb = ks*16 + ((r < 4) ? 0 : 8) + hiA*4 + (r & 3);  // ISA B layout
                bu0[r] = uB[wn*32 +      lmA][pb];
                bu1[r] = uB[wn*32 + 16 + lmA][pb];
            }
            v16bf a0 = __builtin_bit_cast(v16bf, au0);
            v16bf a1 = __builtin_bit_cast(v16bf, au1);
            v16bf b0 = __builtin_bit_cast(v16bf, bu0);
            v16bf b1 = __builtin_bit_cast(v16bf, bu1);

            c00 = __builtin_amdgcn_wmma_f32_16x16x32_bf16(false, a0, false, b0, (short)0, c00, false, false);
            c01 = __builtin_amdgcn_wmma_f32_16x16x32_bf16(false, a0, false, b1, (short)0, c01, false, false);
            c10 = __builtin_amdgcn_wmma_f32_16x16x32_bf16(false, a1, false, b0, (short)0, c10, false, false);
            c11 = __builtin_amdgcn_wmma_f32_16x16x32_bf16(false, a1, false, b1, (short)0, c11, false, false);
        }
        __syncthreads();
    }

    // ---- epilogue: C/D layout -> M = r + 8*laneHi, N = lane%16 ----
    const int nloc  = lmA;
    const int mbase = hiA * 8;
    auto store_tile = [&](v8f cc, int mt, int nt) {
        int gn = n0 + wn*32 + nt*16 + nloc;
        if (gn >= N) return;
        float bv = bias ? bias[gn] : 0.0f;
        int gmb = m0 + wm*32 + mt*16 + mbase;
        #pragma unroll
        for (int r = 0; r < 8; ++r) {
            int gm = gmb + r;
            if (gm >= M) continue;
            float v = cc[r] * scale;
            size_t off = (size_t)gm * ldc + gn;
            if (accumulate) {
                C[off] += v;
            } else {
                v += bv;
                if (act == 1) v = gelu_exact(v);
                else if (act == 2) v = silu_f(v);
                C[off] = v;
            }
        }
    };
    store_tile(c00, 0, 0);
    store_tile(c01, 0, 1);
    store_tile(c10, 1, 0);
    store_tile(c11, 1, 1);
}

// ---------------------------------------------------------------------------
// RMSNorm: one block per row
// ---------------------------------------------------------------------------
__global__ __launch_bounds__(256) void rms_kernel(const float* __restrict__ x,
                                                  float* __restrict__ y,
                                                  const float* __restrict__ w, int H)
{
    __shared__ float red[256];
    const float* xr = x + (size_t)blockIdx.x * H;
    float* yr = y + (size_t)blockIdx.x * H;
    float s = 0.0f;
    for (int d = threadIdx.x; d < H; d += 256) { float v = xr[d]; s += v*v; }
    red[threadIdx.x] = s; __syncthreads();
    for (int st = 128; st > 0; st >>= 1) {
        if (threadIdx.x < st) red[threadIdx.x] += red[threadIdx.x + st];
        __syncthreads();
    }
    float inv = rsqrtf(red[0] / (float)H + 1e-6f);
    for (int d = threadIdx.x; d < H; d += 256) yr[d] = xr[d] * inv * w[d];
}

// ---------------------------------------------------------------------------
// LayerNorm (encoder): one block per row
// ---------------------------------------------------------------------------
__global__ __launch_bounds__(256) void ln_kernel(const float* __restrict__ x,
                                                 float* __restrict__ y,
                                                 const float* __restrict__ w,
                                                 const float* __restrict__ b, int H)
{
    __shared__ float r1[256];
    __shared__ float r2[256];
    const float* xr = x + (size_t)blockIdx.x * H;
    float* yr = y + (size_t)blockIdx.x * H;
    float s1 = 0.0f, s2 = 0.0f;
    for (int d = threadIdx.x; d < H; d += 256) { float v = xr[d]; s1 += v; s2 += v*v; }
    r1[threadIdx.x] = s1; r2[threadIdx.x] = s2; __syncthreads();
    for (int st = 128; st > 0; st >>= 1) {
        if (threadIdx.x < st) { r1[threadIdx.x] += r1[threadIdx.x+st]; r2[threadIdx.x] += r2[threadIdx.x+st]; }
        __syncthreads();
    }
    float mu  = r1[0] / (float)H;
    float var = r2[0] / (float)H - mu*mu;
    float inv = rsqrtf(var + 1e-5f);
    for (int d = threadIdx.x; d < H; d += 256) yr[d] = (xr[d] - mu) * inv * w[d] + b[d];
}

// ---------------------------------------------------------------------------
// Positions + attention mask (serial per batch, trivial work)
// ---------------------------------------------------------------------------
__global__ void pos_amask_kernel(const int* __restrict__ ids,
                                 const float* __restrict__ mmask,
                                 int* __restrict__ pos, float* __restrict__ am)
{
    int b = threadIdx.x;
    if (b >= B_) return;
    int cum = 0;
    for (int t = 0; t < LT_; ++t) {
        int tm = (ids[b*LT_ + t] != 0) ? 1 : 0;
        cum += tm;
        int pv = cum - 1; if (pv < 0) pv = 0;
        pos[b*S_ + t] = pv;
        am[b*S_ + t]  = (float)tm;
    }
    int last = cum - 1; if (last < 0) last = 0;   // max(text_pos): cumsum is monotone
    int sp = last + 1;
    pos[b*S_ + LT_] = sp; am[b*S_ + LT_] = 1.0f;
    for (int t = 1; t < T_; ++t) {
        pos[b*S_ + LT_ + t] = sp + t;
        am[b*S_ + LT_ + t]  = mmask[b*T_ + (t-1)];
    }
}

// ---------------------------------------------------------------------------
// Build input sequence x = [emb[ids] | start | motion_embeds[:, :-1]]
// ---------------------------------------------------------------------------
__global__ void build_x_kernel(const int* __restrict__ ids, const float* __restrict__ emb,
                               const float* __restrict__ start, const float* __restrict__ motemb,
                               float* __restrict__ x)
{
    int idx = blockIdx.x * blockDim.x + threadIdx.x;
    if (idx >= M_ * H_) return;
    int d = idx % H_;
    int s = (idx / H_) % S_;
    int b = idx / (H_ * S_);
    float v;
    if (s < LT_)       v = emb[(size_t)ids[b*LT_ + s] * H_ + d];
    else if (s == LT_) v = start[d];
    else               v = motemb[((size_t)(b*T_ + (s - LT_ - 1))) * H_ + d];
    x[idx] = v;
}

// ---------------------------------------------------------------------------
// RoPE (in-place on (B,S,NH,HD) = (M,1024) layout)
// ---------------------------------------------------------------------------
__global__ void rope_kernel(float* __restrict__ x, const int* __restrict__ pos)
{
    int idx = blockIdx.x * blockDim.x + threadIdx.x;
    if (idx >= M_ * NH_ * 32) return;
    int d2 = idx & 31;
    int hh = (idx >> 5) & 15;
    int s  = (idx >> 9) % S_;
    int b  = idx / (512 * S_);
    float p   = (float)pos[b*S_ + s];
    float inv = powf(10000.0f, -((float)d2) / 32.0f);
    float ang = p * inv;
    float c = cosf(ang), sn = sinf(ang);
    float* base = x + ((size_t)(b*S_ + s)) * H_ + hh * HD_;
    float x1 = base[d2], x2 = base[d2 + 32];
    base[d2]      = x1 * c - x2 * sn;
    base[d2 + 32] = x2 * c + x1 * sn;
}

// ---------------------------------------------------------------------------
// Attention: one block per (b, h, i) row. Scores -> softmax -> weighted V.
// ---------------------------------------------------------------------------
__global__ __launch_bounds__(256) void attn_kernel(const float* __restrict__ q,
                                                   const float* __restrict__ k,
                                                   const float* __restrict__ v,
                                                   float* __restrict__ o,
                                                   const float* __restrict__ amask)
{
    __shared__ float sQ[HD_];
    __shared__ float sP[S_];
    __shared__ float red[256];

    int idx = blockIdx.x;
    int i = idx % S_;
    int h = (idx / S_) % NH_;
    int b = idx / (S_ * NH_);
    int tid = threadIdx.x;

    if (tid < HD_) sQ[tid] = q[((size_t)(b*S_ + i)) * H_ + h*HD_ + tid];
    __syncthreads();

    for (int j = tid; j < S_; j += 256) {
        float s;
        if (j <= i && amask[b*S_ + j] > 0.0f) {
            const float* kr = k + ((size_t)(b*S_ + j)) * H_ + h*HD_;
            float acc = 0.0f;
            #pragma unroll 8
            for (int d = 0; d < HD_; ++d) acc = fmaf(sQ[d], kr[d], acc);
            s = acc * 0.125f;                 // 1/sqrt(64)
        } else {
            s = -1.0e9f;
        }
        sP[j] = s;
    }
    __syncthreads();

    // max reduce
    float lm = -1.0e30f;
    for (int j = tid; j < S_; j += 256) lm = fmaxf(lm, sP[j]);
    red[tid] = lm; __syncthreads();
    for (int st = 128; st > 0; st >>= 1) {
        if (tid < st) red[tid] = fmaxf(red[tid], red[tid + st]);
        __syncthreads();
    }
    float mx = red[0]; __syncthreads();

    // exp + sum
    float ls = 0.0f;
    for (int j = tid; j < S_; j += 256) { float e = expf(sP[j] - mx); sP[j] = e; ls += e; }
    red[tid] = ls; __syncthreads();
    for (int st = 128; st > 0; st >>= 1) {
        if (tid < st) red[tid] += red[tid + st];
        __syncthreads();
    }
    float Z = red[0]; __syncthreads();

    if (tid < HD_) {
        float acc = 0.0f;
        const float* vb = v + (size_t)b * S_ * H_ + h*HD_ + tid;
        for (int j = 0; j < S_; ++j) acc = fmaf(sP[j], vb[(size_t)j * H_], acc);
        o[((size_t)(b*S_ + i)) * H_ + h*HD_ + tid] = acc / Z;
    }
}

// ---------------------------------------------------------------------------
// g = silu(g1) * g2  (in-place into g1)
// ---------------------------------------------------------------------------
__global__ void silu_mul_kernel(float* __restrict__ g1, const float* __restrict__ g2, int n)
{
    int idx = blockIdx.x * blockDim.x + threadIdx.x;
    if (idx >= n) return;
    g1[idx] = silu_f(g1[idx]) * g2[idx];
}

// ---------------------------------------------------------------------------
// Gather rows: dst[(b*npb+t)] = src[(b*S_ + base + t)]
// ---------------------------------------------------------------------------
__global__ void gather_rows_kernel(const float* __restrict__ src, float* __restrict__ dst,
                                   int npb, int base, int ncols)
{
    int idx = blockIdx.x * blockDim.x + threadIdx.x;
    if (idx >= B_ * npb * ncols) return;
    int c = idx % ncols;
    int t = (idx / ncols) % npb;
    int b = idx / (ncols * npb);
    dst[idx] = src[((size_t)(b*S_ + base + t)) * ncols + c];
}

// ---------------------------------------------------------------------------
// Language loss: per row log-softmax + NLL at label  (252 blocks)
// ---------------------------------------------------------------------------
__global__ __launch_bounds__(256) void lang_loss_kernel(const float* __restrict__ logits,
                                                        const int* __restrict__ ids,
                                                        float* __restrict__ nll_v,
                                                        float* __restrict__ lm_v)
{
    __shared__ float red[256];
    int r = blockIdx.x;                 // 0..251
    int b = r / (LT_ - 1), t = r % (LT_ - 1);
    const float* lr = logits + (size_t)r * V_;
    int tid = threadIdx.x;

    float lm = -1.0e30f;
    for (int j = tid; j < V_; j += 256) lm = fmaxf(lm, lr[j]);
    red[tid] = lm; __syncthreads();
    for (int st = 128; st > 0; st >>= 1) {
        if (tid < st) red[tid] = fmaxf(red[tid], red[tid + st]);
        __syncthreads();
    }
    float mx = red[0]; __syncthreads();

    float ls = 0.0f;
    for (int j = tid; j < V_; j += 256) ls += expf(lr[j] - mx);
    red[tid] = ls; __syncthreads();
    for (int st = 128; st > 0; st >>= 1) {
        if (tid < st) red[tid] += red[tid + st];
        __syncthreads();
    }
    float Z = red[0];

    if (tid == 0) {
        int lab = ids[b*LT_ + t + 1];
        float logp = lr[lab] - mx - logf(Z);
        float msk = (lab != 0) ? 1.0f : 0.0f;
        nll_v[r] = -logp * msk;
        lm_v[r]  = msk;
    }
}

// ---------------------------------------------------------------------------
// Motion losses: one thread per (b,t); writes per-row partials.
// lossbuf layout: pos | mm | vel | vm | fc | cont (each BT_ floats)
// ---------------------------------------------------------------------------
__global__ void motion_loss_kernel(const float* __restrict__ pred,
                                   const float* __restrict__ motion,
                                   const float* __restrict__ mmask,
                                   const float* __restrict__ meanv,
                                   const float* __restrict__ stdv,
                                   float* __restrict__ lossbuf)
{
    int idx = blockIdx.x * blockDim.x + threadIdx.x;
    if (idx >= BT_) return;
    int t = idx % T_;
    const float* pr = pred   + (size_t)idx * D_;
    const float* mo = motion + (size_t)idx * D_;
    float mm = mmask[idx];

    // loss_pos: mean sq err over D
    float s = 0.0f;
    for (int d = 0; d < D_; ++d) { float df = pr[d] - mo[d]; s += df*df; }
    lossbuf[0*BT_ + idx] = (s / (float)D_) * mm;
    lossbuf[1*BT_ + idx] = mm;

    // loss_contact (BCE over 4 contact channels)
    float bce = 0.0f;
    for (int j = 0; j < 4; ++j) {
        float cp = pr[259 + j] * stdv[259 + j] + meanv[259 + j];
        float cl = (cp - 0.5f) * 10.0f;
        float gm = mo[259 + j] * stdv[259 + j] + meanv[259 + j];
        float gt = (gm > 0.5f) ? 1.0f : 0.0f;
        bce += softplus_f(cl) - cl * gt;
    }
    lossbuf[5*BT_ + idx] = (bce * 0.25f) * mm;

    // velocity + foot-contact losses (t >= 1)
    if (t == 0) {
        lossbuf[2*BT_ + idx] = 0.0f;
        lossbuf[3*BT_ + idx] = 0.0f;
        lossbuf[4*BT_ + idx] = 0.0f;
        return;
    }
    const float* pr1 = pr - D_;
    const float* mo1 = mo - D_;
    float vm = mm;                       // vmask = motion_mask[:,1:] at this t
    float sv = 0.0f;
    for (int d = 0; d < D_; ++d) {
        float pv = pr[d] - pr1[d];
        float tv = mo[d] - mo1[d];
        float df = pv - tv;
        sv += df*df;
    }
    lossbuf[2*BT_ + idx] = (sv / (float)D_) * vm;
    lossbuf[3*BT_ + idx] = vm;

    float dl = 0.0f, dr = 0.0f;
    for (int j = 0; j < 3; ++j) {
        float a = pr[157 + j] * stdv[157 + j] + meanv[157 + j];
        float c = pr1[157 + j] * stdv[157 + j] + meanv[157 + j];
        dl += (a - c) * (a - c);
        float a2 = pr[160 + j] * stdv[160 + j] + meanv[160 + j];
        float c2 = pr1[160 + j] * stdv[160 + j] + meanv[160 + j];
        dr += (a2 - c2) * (a2 - c2);
    }
    float vl = sqrtf(dl), vr = sqrtf(dr);
    float cp0 = sigmoid_f((pr[259] * stdv[259] + meanv[259] - 0.5f) * 10.0f);
    float cp1 = sigmoid_f((pr[260] * stdv[260] + meanv[260] - 0.5f) * 10.0f);
    float cp2 = sigmoid_f((pr[261] * stdv[261] + meanv[261] - 0.5f) * 10.0f);
    float cp3 = sigmoid_f((pr[262] * stdv[262] + meanv[262] - 0.5f) * 10.0f);
    float cl = fmaxf(cp0, cp1), cr = fmaxf(cp2, cp3);
    lossbuf[4*BT_ + idx] = (vl * cl + vr * cr) * vm;
}

// ---------------------------------------------------------------------------
// Final deterministic reduction + total loss
// ---------------------------------------------------------------------------
__global__ __launch_bounds__(256) void finalize_kernel(const float* __restrict__ lossbuf,
                                                       const float* __restrict__ nll_v,
                                                       const float* __restrict__ lm_v,
                                                       float* __restrict__ out0)
{
    __shared__ float red[256];
    int tid = threadIdx.x;
    auto rsum = [&](const float* a, int n) {
        float s = 0.0f;
        for (int j = tid; j < n; j += 256) s += a[j];
        red[tid] = s; __syncthreads();
        for (int st = 128; st > 0; st >>= 1) {
            if (tid < st) red[tid] += red[tid + st];
            __syncthreads();
        }
        float r = red[0]; __syncthreads();
        return r;
    };
    float Spos  = rsum(lossbuf + 0*BT_, BT_);
    float Smm   = rsum(lossbuf + 1*BT_, BT_);
    float Svel  = rsum(lossbuf + 2*BT_, BT_);
    float Svm   = rsum(lossbuf + 3*BT_, BT_);
    float Sfc   = rsum(lossbuf + 4*BT_, BT_);
    float Scont = rsum(lossbuf + 5*BT_, BT_);
    float Snll  = rsum(nll_v, TLROWS_);
    float Slm   = rsum(lm_v, TLROWS_);
    if (tid == 0) {
        float lpos  = Spos  / (Smm + 1e-8f);
        float lvel  = Svel  / (Svm + 1e-8f);
        float lfc   = Sfc   / (Svm + 1e-8f);
        float lcont = Scont / (Smm + 1e-8f);
        float llang = Snll  / fmaxf(Slm, 1.0f);
        out0[0] = 1.0f*lpos + 0.5f*lvel + 0.5f*lfc + 0.1f*lcont + 0.1f*llang;
    }
}

// ---------------------------------------------------------------------------
// Host side
// ---------------------------------------------------------------------------
extern "C" void kernel_launch(void* const* d_in, const int* in_sizes, int n_in,
                              void* d_out, int out_size, void* d_ws, size_t ws_size,
                              hipStream_t stream)
{
    const float* emb   = (const float*)d_in[0];
    const float* Wq    = (const float*)d_in[1];
    const float* Wk    = (const float*)d_in[2];
    const float* Wv    = (const float*)d_in[3];
    const float* Wo    = (const float*)d_in[4];
    const float* Wg    = (const float*)d_in[5];
    const float* Wu    = (const float*)d_in[6];
    const float* Wd    = (const float*)d_in[7];
    const float* Aq    = (const float*)d_in[8];
    const float* Bq    = (const float*)d_in[9];
    const float* Ak    = (const float*)d_in[10];
    const float* Bk    = (const float*)d_in[11];
    const float* Av    = (const float*)d_in[12];
    const float* Bv    = (const float*)d_in[13];
    const float* Ao    = (const float*)d_in[14];
    const float* Bo    = (const float*)d_in[15];
    const float* Ag    = (const float*)d_in[16];
    const float* Bg    = (const float*)d_in[17];
    const float* Au    = (const float*)d_in[18];
    const float* Bu    = (const float*)d_in[19];
    const float* Ad    = (const float*)d_in[20];
    const float* Bd    = (const float*)d_in[21];
    const float* ln1   = (const float*)d_in[22];
    const float* ln2   = (const float*)d_in[23];
    const float* lnf   = (const float*)d_in[24];
    const float* eW1   = (const float*)d_in[25];
    const float* eb1   = (const float*)d_in[26];
    const float* eW2   = (const float*)d_in[27];
    const float* eb2   = (const float*)d_in[28];
    const float* eW3   = (const float*)d_in[29];
    const float* eb3   = (const float*)d_in[30];
    const float* elnw  = (const float*)d_in[31];
    const float* elnb  = (const float*)d_in[32];
    const float* dW1   = (const float*)d_in[33];
    const float* db1   = (const float*)d_in[34];
    const float* dW2   = (const float*)d_in[35];
    const float* db2   = (const float*)d_in[36];
    const float* dW3   = (const float*)d_in[37];
    const float* db3   = (const float*)d_in[38];
    const float* start = (const float*)d_in[39];
    const float* meanv = (const float*)d_in[40];
    const float* stdv  = (const float*)d_in[41];
    const float* motion= (const float*)d_in[42];
    const int*   ids   = (const int*)d_in[43];
    const float* mmask = (const float*)d_in[44];

    float* out = (float*)d_out;            // [0] = total, [1..] = pred (B,T,D)
    float* pred = out + 1;
    float* ws = (float*)d_ws;

    // workspace layout (float elements)
    const size_t SZR   = (size_t)M_ * H_;          // 1040*1024
    const size_t OFF_H   = 0;
    const size_t OFF_HN  = 1*SZR;
    const size_t OFF_Q   = 2*SZR;
    const size_t OFF_K   = 3*SZR;
    const size_t OFF_V   = 4*SZR;
    const size_t OFF_O   = 5*SZR;
    const size_t OFF_G1  = 6*SZR;                  // 1040*4096
    const size_t OFF_G2  = OFF_G1 + (size_t)M_*F_;
    const size_t OFF_TMP = OFF_G2 + (size_t)M_*F_; // 1040*32
    const size_t OFF_POS = OFF_TMP + (size_t)M_*R_;
    const size_t OFF_AM  = OFF_POS + M_;
    const size_t OFF_LB  = OFF_AM + M_;            // loss partials: 6*BT_
    const size_t OFF_NLL = OFF_LB + 6*(size_t)BT_;
    const size_t OFF_LM  = OFF_NLL + TLROWS_;
    // aliases (lifetimes disjoint):
    float* buf_h   = ws + OFF_H;
    float* buf_hn  = ws + OFF_HN;
    float* buf_q   = ws + OFF_Q;
    float* buf_k   = ws + OFF_K;
    float* buf_v   = ws + OFF_V;
    float* buf_o   = ws + OFF_O;
    float* buf_g1  = ws + OFF_G1;
    float* buf_g2  = ws + OFF_G2;
    float* buf_tmp = ws + OFF_TMP;
    int*   pos_i   = (int*)(ws + OFF_POS);
    float* am      = ws + OFF_AM;
    float* lossbuf = ws + OFF_LB;
    float* nll_v   = ws + OFF_NLL;
    float* lm_v    = ws + OFF_LM;
    float* enc1    = buf_q;       // 784x1024, free before layers
    float* enc2    = buf_k;
    float* motemb  = buf_o;       // motion_embeds, free after build_x
    float* logits  = buf_g1;      // 252*32000 <= 2*1040*4096, free after layers
    float* arows   = buf_o;       // 252x1024
    float* decin   = buf_q;       // 784x1024
    float* dec1    = buf_k;       // 784x1024
    float* dec2    = buf_v;       // 784x263

    auto gemm = [&](const float* A, const float* W, float* C, int M, int N, int K,
                    int lda, int ldc, const float* bias, float scale, int acc, int act) {
        dim3 g((N + GTN - 1) / GTN, (M + GTM - 1) / GTM);
        gemm_bf16_kernel<<<g, 256, 0, stream>>>(A, W, C, M, N, K, lda, K, ldc,
                                                bias, scale, acc, act);
    };

    // ---------------- motion encoder ----------------
    gemm(motion, eW1, enc1, BT_, H_, D_, D_, H_, eb1, 1.0f, 0, 1);   // gelu
    gemm(enc1,   eW2, enc2, BT_, H_, H_, H_, H_, eb2, 1.0f, 0, 1);   // gelu
    gemm(enc2,   eW3, enc1, BT_, H_, H_, H_, H_, eb3, 1.0f, 0, 0);
    ln_kernel<<<BT_, 256, 0, stream>>>(enc1, motemb, elnw, elnb, H_);

    // ---------------- positions / mask / input sequence ----------------
    pos_amask_kernel<<<1, 32, 0, stream>>>(ids, mmask, pos_i, am);
    build_x_kernel<<<(M_*H_ + 255)/256, 256, 0, stream>>>(ids, emb, start, motemb, buf_h);

    // ---------------- transformer layers ----------------
    for (int i = 0; i < NL_; ++i) {
        const float* wq = Wq + (size_t)i*H_*H_;
        const float* wk = Wk + (size_t)i*H_*H_;
        const float* wv = Wv + (size_t)i*H_*H_;
        const float* wo = Wo + (size_t)i*H_*H_;
        const float* wg = Wg + (size_t)i*F_*H_;
        const float* wu = Wu + (size_t)i*F_*H_;
        const float* wd = Wd + (size_t)i*H_*F_;
        const float* aq = Aq + (size_t)i*R_*H_;  const float* bq = Bq + (size_t)i*H_*R_;
        const float* ak = Ak + (size_t)i*R_*H_;  const float* bk = Bk + (size_t)i*H_*R_;
        const float* av = Av + (size_t)i*R_*H_;  const float* bv = Bv + (size_t)i*H_*R_;
        const float* ao = Ao + (size_t)i*R_*H_;  const float* bo = Bo + (size_t)i*H_*R_;
        const float* ag = Ag + (size_t)i*R_*H_;  const float* bg = Bg + (size_t)i*F_*R_;
        const float* au = Au + (size_t)i*R_*H_;  const float* bu = Bu + (size_t)i*F_*R_;
        const float* ad = Ad + (size_t)i*R_*F_;  const float* bd = Bd + (size_t)i*H_*R_;

        rms_kernel<<<M_, 256, 0, stream>>>(buf_h, buf_hn, ln1 + i*H_, H_);

        // q / k / v projections (W + LoRA)
        gemm(buf_hn, wq, buf_q, M_, H_, H_, H_, H_, nullptr, 1.0f, 0, 0);
        gemm(buf_hn, aq, buf_tmp, M_, R_, H_, H_, R_, nullptr, 1.0f, 0, 0);
        gemm(buf_tmp, bq, buf_q, M_, H_, R_, R_, H_, nullptr, LORA_SCALE_, 1, 0);

        gemm(buf_hn, wk, buf_k, M_, H_, H_, H_, H_, nullptr, 1.0f, 0, 0);
        gemm(buf_hn, ak, buf_tmp, M_, R_, H_, H_, R_, nullptr, 1.0f, 0, 0);
        gemm(buf_tmp, bk, buf_k, M_, H_, R_, R_, H_, nullptr, LORA_SCALE_, 1, 0);

        gemm(buf_hn, wv, buf_v, M_, H_, H_, H_, H_, nullptr, 1.0f, 0, 0);
        gemm(buf_hn, av, buf_tmp, M_, R_, H_, H_, R_, nullptr, 1.0f, 0, 0);
        gemm(buf_tmp, bv, buf_v, M_, H_, R_, R_, H_, nullptr, LORA_SCALE_, 1, 0);

        rope_kernel<<<(M_*NH_*32 + 255)/256, 256, 0, stream>>>(buf_q, pos_i);
        rope_kernel<<<(M_*NH_*32 + 255)/256, 256, 0, stream>>>(buf_k, pos_i);

        attn_kernel<<<B_*NH_*S_, 256, 0, stream>>>(buf_q, buf_k, buf_v, buf_o, am);

        // h += lora(o, Wo)
        gemm(buf_o, wo, buf_h, M_, H_, H_, H_, H_, nullptr, 1.0f, 1, 0);
        gemm(buf_o, ao, buf_tmp, M_, R_, H_, H_, R_, nullptr, 1.0f, 0, 0);
        gemm(buf_tmp, bo, buf_h, M_, H_, R_, R_, H_, nullptr, LORA_SCALE_, 1, 0);

        // MLP
        rms_kernel<<<M_, 256, 0, stream>>>(buf_h, buf_hn, ln2 + i*H_, H_);

        gemm(buf_hn, wg, buf_g1, M_, F_, H_, H_, F_, nullptr, 1.0f, 0, 0);
        gemm(buf_hn, ag, buf_tmp, M_, R_, H_, H_, R_, nullptr, 1.0f, 0, 0);
        gemm(buf_tmp, bg, buf_g1, M_, F_, R_, R_, F_, nullptr, LORA_SCALE_, 1, 0);

        gemm(buf_hn, wu, buf_g2, M_, F_, H_, H_, F_, nullptr, 1.0f, 0, 0);
        gemm(buf_hn, au, buf_tmp, M_, R_, H_, H_, R_, nullptr, 1.0f, 0, 0);
        gemm(buf_tmp, bu, buf_g2, M_, F_, R_, R_, F_, nullptr, LORA_SCALE_, 1, 0);

        silu_mul_kernel<<<((size_t)M_*F_ + 255)/256, 256, 0, stream>>>(buf_g1, buf_g2, M_*F_);

        gemm(buf_g1, wd, buf_h, M_, H_, F_, F_, H_, nullptr, 1.0f, 1, 0);
        gemm(buf_g1, ad, buf_tmp, M_, R_, F_, F_, R_, nullptr, 1.0f, 0, 0);
        gemm(buf_tmp, bd, buf_h, M_, H_, R_, R_, H_, nullptr, LORA_SCALE_, 1, 0);
    }

    // ---------------- final norm ----------------
    rms_kernel<<<M_, 256, 0, stream>>>(buf_h, buf_hn, lnf, H_);

    // ---------------- language loss (only first LT-1 positions needed) ----
    gather_rows_kernel<<<(TLROWS_*H_ + 255)/256, 256, 0, stream>>>(buf_hn, arows, LT_-1, 0, H_);
    gemm(arows, emb, logits, TLROWS_, V_, H_, H_, V_, nullptr, 1.0f, 0, 0);
    lang_loss_kernel<<<TLROWS_, 256, 0, stream>>>(logits, ids, nll_v, lm_v);

    // ---------------- motion decoder ----------------
    gather_rows_kernel<<<(BT_*H_ + 255)/256, 256, 0, stream>>>(buf_hn, decin, T_, LT_, H_);
    gemm(decin, dW1, dec1, BT_, H_, H_, H_, H_, db1, 1.0f, 0, 1);    // gelu
    gemm(dec1,  dW2, dec2, BT_, D_, H_, H_, D_, db2, 1.0f, 0, 1);    // gelu
    gemm(dec2,  dW3, pred, BT_, D_, D_, D_, D_, db3, 1.0f, 0, 0);

    // ---------------- losses ----------------
    motion_loss_kernel<<<(BT_ + 255)/256, 256, 0, stream>>>(pred, motion, mmask,
                                                            meanv, stdv, lossbuf);
    finalize_kernel<<<1, 256, 0, stream>>>(lossbuf, nll_v, lm_v, out);
}
